// RelPosAttentionMethod3_53712861004255
// MI455X (gfx1250) — matrix operand order repaired and checked by reading, not compiled, not run on hardware
//
#include <hip/hip_runtime.h>
#include <hip/hip_bf16.h>

// ---------------------------------------------------------------------------
// RelPos attention for MI455X (gfx1250):
//  - all GEMMs via v_wmma_f32_16x16x32_bf16 (fp32 accumulate)
//  - double-buffered LDS tiles fed by async global->LDS (ASYNCcnt) when the
//    toolchain exposes the gfx1250 builtins, scalar ds_store fallback otherwise
// ---------------------------------------------------------------------------

typedef __attribute__((ext_vector_type(16))) __bf16 v16bf;
typedef __attribute__((ext_vector_type(8)))  float  v8f;
typedef int vi4 __attribute__((vector_size(16)));   // int4 for async-LDS builtins

#define D_MODEL 2048
#define NH      8
#define DH      256
#define LSEQ    256
#define BATCH   8
#define BL      (BATCH * LSEQ)   // 2048 rows of X
#define TPAD    40               // LDS pitch (bf16): 80B rows -> 16B-aligned chunks

#if __has_builtin(__builtin_amdgcn_global_load_async_to_lds_b128)
#define HAVE_ASYNC_LDS 1
#endif

// ---- async (or fallback) 32-byte global -> LDS copy -----------------------

__device__ inline void cp32B(const __bf16* __restrict__ g, __bf16* s) {
#if defined(HAVE_ASYNC_LDS)
  // Param types per hipcc diagnostic: (int4 __device__*, int4 __shared__*, Ii, Ii)
  __builtin_amdgcn_global_load_async_to_lds_b128(
      (__attribute__((address_space(1))) vi4*)g,
      (__attribute__((address_space(3))) vi4*)s, 0, 0);
  __builtin_amdgcn_global_load_async_to_lds_b128(
      (__attribute__((address_space(1))) vi4*)(g + 8),
      (__attribute__((address_space(3))) vi4*)(s + 8), 0, 0);
#else
#pragma unroll
  for (int i = 0; i < 16; ++i) s[i] = g[i];
#endif
}

__device__ inline void wait_async_lds() {
#if defined(HAVE_ASYNC_LDS)
#if __has_builtin(__builtin_amdgcn_s_wait_asynccnt)
  __builtin_amdgcn_s_wait_asynccnt(0);
#else
  asm volatile("s_wait_asynccnt 0" ::: "memory");
#endif
#endif
}

// ---- WMMA fragment helpers (layouts per CDNA5 ISA 7.12.2) -----------------

// A fragment, 16x32 bf16. Lanes 0-15: row M=lane, K 0..7 (elems 0..7) and
// 16..23 (elems 8..15). Lanes 16-31: row M=lane-16, K 8..15 and 24..31.
__device__ inline v16bf frag_a_16x32(const __bf16* tile, int ldm, int lane) {
  const int m  = lane & 15;
  const int kh = (lane >> 4) & 1;
  const __bf16* p = tile + m * ldm;
  v16bf a;
#pragma unroll
  for (int i = 0; i < 8; ++i) {
    a[i]     = p[kh * 8 + i];
    a[i + 8] = p[16 + kh * 8 + i];
  }
  return a;
}

// B fragment, 32x16 bf16, sourced from a B^T [n][k] LDS tile:
// lanes 0-15 hold K 0..7 and 16..23 of column N=lane; lanes 16-31 hold K 8..15 and 24..31.
__device__ inline v16bf frag_bT_32x16(const __bf16* tileT, int ldn, int lane) {
  const int n  = lane & 15;
  const int kh = (lane >> 4) & 1;
  const __bf16* p = tileT + n * ldn;
  v16bf b;
#pragma unroll
  for (int i = 0; i < 8; ++i) {
    b[i]     = p[kh * 8 + i];
    b[i + 8] = p[16 + kh * 8 + i];
  }
  return b;
}

__device__ inline v8f wmma_bf16(v16bf a, v16bf b, v8f c) {
  return __builtin_amdgcn_wmma_f32_16x16x32_bf16(
      /*neg_a=*/false, a, /*neg_b=*/false, b,
      /*c_mod=*/(short)0, c, /*reuse_a=*/false, /*reuse_b=*/false);
}

__device__ inline v8f v8f_zero() {
  v8f z = {0.f, 0.f, 0.f, 0.f, 0.f, 0.f, 0.f, 0.f};
  return z;
}

// ---- 1. fp32 -> bf16 conversion -------------------------------------------

__global__ void __launch_bounds__(256)
cvt_f32_to_bf16(const float* __restrict__ src, __bf16* __restrict__ dst, int n) {
  int i = blockIdx.x * blockDim.x + threadIdx.x;
  const int stride = gridDim.x * blockDim.x;
  for (; i < n; i += stride) dst[i] = (__bf16)src[i];
}

// ---- 2./3./7. Unified pipelined GEMM: C = A @ B (B supplied as B^T) -------
// A:  row-major [128-tile rows][ld], contiguous along K.
// Bt: row-major [n][k] (== B^T), contiguous along K.
//   * projections: Bt = torch Linear weight W (W[n,k]) -> no transpose
//   * content scores: Bt = k[b,h] (k[m,d]) -> no transpose
// mode 0: bf16 out, (B,H,L,Dh) head-major permutation (q/k/v), +bias
// mode 1: fp32 out, row-major D_MODEL (final projection), +bias
// mode 2: fp32 out, scores[z][row][col] (content)
__global__ void __launch_bounds__(256)
gemm_bt(const __bf16* __restrict__ A, unsigned aStrideZ,
        const __bf16* __restrict__ Bt, unsigned bStrideZ,
        int ld, int kTotal, const float* __restrict__ bias, int mode,
        __bf16* __restrict__ outB, float* __restrict__ outF) {
  __shared__ __bf16 sA[2][128 * TPAD];
  __shared__ __bf16 sB[2][128 * TPAD];
  const int z = blockIdx.z;
  const __bf16* Az = A + (size_t)z * aStrideZ;
  const __bf16* Bz = Bt + (size_t)z * bStrideZ;
  const int m0 = blockIdx.x * 128, n0 = blockIdx.y * 128;
  const int t = threadIdx.x, lane = t & 31, wid = t >> 5;
  const int wm = wid & 1, wn = wid >> 1;          // 2x4 wave grid
  const int srow = t >> 1;                         // 0..127
  const int scol = (t & 1) * 16;                   // 0 or 16

  v8f acc[4][2];
#pragma unroll
  for (int i = 0; i < 4; ++i)
#pragma unroll
    for (int j = 0; j < 2; ++j) acc[i][j] = v8f_zero();

  auto stage = [&](int k0, int buf) {
    cp32B(Az + (size_t)(m0 + srow) * ld + k0 + scol, &sA[buf][srow * TPAD + scol]);
    cp32B(Bz + (size_t)(n0 + srow) * ld + k0 + scol, &sB[buf][srow * TPAD + scol]);
  };

  stage(0, 0);
  wait_async_lds();
  __syncthreads();
  int buf = 0;
  for (int k0 = 0; k0 < kTotal; k0 += 32, buf ^= 1) {
    if (k0 + 32 < kTotal) stage(k0 + 32, buf ^ 1);  // prefetch next tile
    v16bf bfr[2];
#pragma unroll
    for (int ns = 0; ns < 2; ++ns)
      bfr[ns] = frag_bT_32x16(&sB[buf][(wn * 32 + ns * 16) * TPAD], TPAD, lane);
#pragma unroll
    for (int ms = 0; ms < 4; ++ms) {
      v16bf afr = frag_a_16x32(&sA[buf][(wm * 64 + ms * 16) * TPAD], TPAD, lane);
#pragma unroll
      for (int ns = 0; ns < 2; ++ns)
        acc[ms][ns] = wmma_bf16(afr, bfr[ns], acc[ms][ns]);
    }
    wait_async_lds();
    __syncthreads();
  }

  const int colLane = lane & 15;
  const int rquad = (lane >> 4) * 8;
#pragma unroll
  for (int ms = 0; ms < 4; ++ms) {
#pragma unroll
    for (int ns = 0; ns < 2; ++ns) {
      const int col = n0 + wn * 32 + ns * 16 + colLane;
      const float bv = (mode == 2) ? 0.f : bias[col];
#pragma unroll
      for (int i = 0; i < 8; ++i) {
        const int row = m0 + wm * 64 + ms * 16 + rquad + i;
        const float v = acc[ms][ns][i] + bv;
        if (mode == 0) {
          const int b = row >> 8, l = row & 255;
          const int h = col >> 8, d = col & 255;
          outB[((size_t)((b * NH + h) * LSEQ + l)) * DH + d] = (__bf16)v;
        } else if (mode == 1) {
          outF[(size_t)row * D_MODEL + col] = v;
        } else {
          outF[(size_t)z * LSEQ * LSEQ + (size_t)row * LSEQ + col] = v;
        }
      }
    }
  }
}

// ---- 4. rel scores: S[b,h,l,m] += sum_d q[b,h,l,d]*rel[d-l+255, 256h+m] ---
// One WG per (h,l): an (8x256)@(256x256) GEMM over the batch dim, M padded
// to 16 (rows 8..15 zero). B rows are consecutive rel_table rows starting at
// 255-l, columns [256h, 256h+256). B staging is an inherent transpose ->
// scalar ds stores, single buffer (this is <3% of total FLOPs).
__global__ void __launch_bounds__(128)
attn_rel(const __bf16* __restrict__ Q, const __bf16* __restrict__ RT,
         float* __restrict__ S) {
  __shared__ __bf16 sA[16 * TPAD];
  __shared__ __bf16 sB[256 * TPAD];
  const int hl = blockIdx.x;
  const int h = hl >> 8, l = hl & 255;
  const int t = threadIdx.x, lane = t & 31, wid = t >> 5;  // 4 waves
  v8f acc[4];
#pragma unroll
  for (int i = 0; i < 4; ++i) acc[i] = v8f_zero();

  for (int k0 = 0; k0 < DH; k0 += 32) {
    {  // A tile 16x32: rows 0..7 = q[b,h,l,k0+..], rows 8..15 zero
      int idx = t;
#pragma unroll
      for (int rep = 0; rep < 4; ++rep, idx += 128) {
        const int row = idx >> 5, col = idx & 31;
        __bf16 v = (__bf16)0.0f;
        if (row < BATCH)
          v = Q[((size_t)(row * NH + h) * LSEQ + l) * DH + k0 + col];
        sA[row * TPAD + col] = v;
      }
    }
    {  // B^T tile [m][d]: sB[m][d] = RT[(255-l+k0+d)*D + 256h + m]
      const int d  = t >> 2;           // 0..31
      const int mb = (t & 3) * 64;     // 0,64,128,192
      const __bf16* g = RT + (size_t)(255 - l + k0 + d) * D_MODEL + h * DH + mb;
#pragma unroll
      for (int i = 0; i < 64; ++i) sB[(mb + i) * TPAD + d] = g[i];
    }
    __syncthreads();
    const v16bf afr = frag_a_16x32(sA, TPAD, lane);
#pragma unroll
    for (int ns = 0; ns < 4; ++ns) {
      v16bf bfr = frag_bT_32x16(sB + (wid * 64 + ns * 16) * TPAD, TPAD, lane);
      acc[ns] = wmma_bf16(afr, bfr, acc[ns]);
    }
    __syncthreads();
  }
  // C rows 0..7 (lanes 0..15, elems 0..7) are the valid batch rows.
  if (lane < 16) {
#pragma unroll
    for (int ns = 0; ns < 4; ++ns) {
      const int col = wid * 64 + ns * 16 + lane;
#pragma unroll
      for (int b = 0; b < 8; ++b) {
        const size_t o = ((size_t)((b * NH + h) * LSEQ + l)) * LSEQ + col;
        S[o] += acc[ns][b];
      }
    }
  }
}

// ---- 5. softmax over last axis (256), scale=1/sqrt(256)=1/16 --------------
__global__ void __launch_bounds__(256)
softmax_rows(const float* __restrict__ S, __bf16* __restrict__ P) {
  const int wid = threadIdx.x >> 5, lane = threadIdx.x & 31;
  const int row = blockIdx.x * 8 + wid;
  const float* s = S + (size_t)row * LSEQ;
  float x[8];
#pragma unroll
  for (int j = 0; j < 8; ++j) x[j] = s[lane + 32 * j] * 0.0625f;
  float m = x[0];
#pragma unroll
  for (int j = 1; j < 8; ++j) m = fmaxf(m, x[j]);
#pragma unroll
  for (int off = 16; off > 0; off >>= 1) m = fmaxf(m, __shfl_xor(m, off, 32));
  float e[8], sum = 0.f;
#pragma unroll
  for (int j = 0; j < 8; ++j) { e[j] = __expf(x[j] - m); sum += e[j]; }
#pragma unroll
  for (int off = 16; off > 0; off >>= 1) sum += __shfl_xor(sum, off, 32);
  const float inv = 1.0f / sum;
  __bf16* p = P + (size_t)row * LSEQ;
#pragma unroll
  for (int j = 0; j < 8; ++j) p[lane + 32 * j] = (__bf16)(e[j] * inv);
}

// ---- 6. context: ctx[b,l,256h+d] = sum_m attn[bh,l,m]*v[bh,m,d] -----------
// A (attn) staged with async copies; V tile is transposed during staging
// (coalesced global reads along d, strided scalar LDS writes).
__global__ void __launch_bounds__(256)
attn_ctx(const __bf16* __restrict__ P, const __bf16* __restrict__ V,
         __bf16* __restrict__ ctx) {
  __shared__ __bf16 sA[2][128 * TPAD];
  __shared__ __bf16 sB[2][128 * TPAD];
  const int bh = blockIdx.z;
  const int m0 = blockIdx.x * 128, n0 = blockIdx.y * 128;
  const __bf16* Pb = P + (size_t)bh * LSEQ * LSEQ;
  const __bf16* Vb = V + (size_t)bh * LSEQ * DH;
  const int t = threadIdx.x, lane = t & 31, wid = t >> 5;
  const int wm = wid & 1, wn = wid >> 1;
  const int srow = t >> 1, scol = (t & 1) * 16;
  v8f acc[4][2];
#pragma unroll
  for (int i = 0; i < 4; ++i)
#pragma unroll
    for (int j = 0; j < 2; ++j) acc[i][j] = v8f_zero();

  auto stage = [&](int k0, int buf) {
    // A rows contiguous along m=K -> async copy
    cp32B(Pb + (size_t)(m0 + srow) * LSEQ + k0 + scol,
          &sA[buf][srow * TPAD + scol]);
    // B^T[n=d][k=m] = v[m][d]: transpose during staging
    const int kk = t >> 3;           // 0..31  (m within tile)
    const int nb = (t & 7) * 16;     // 0..112 (d base)
    const __bf16* gV = Vb + (size_t)(k0 + kk) * DH + n0 + nb;
#pragma unroll
    for (int i = 0; i < 16; ++i) sB[buf][(nb + i) * TPAD + kk] = gV[i];
  };

  stage(0, 0);
  wait_async_lds();
  __syncthreads();
  int buf = 0;
  for (int k0 = 0; k0 < LSEQ; k0 += 32, buf ^= 1) {
    if (k0 + 32 < LSEQ) stage(k0 + 32, buf ^ 1);
    v16bf bfr[2];
#pragma unroll
    for (int ns = 0; ns < 2; ++ns)
      bfr[ns] = frag_bT_32x16(&sB[buf][(wn * 32 + ns * 16) * TPAD], TPAD, lane);
#pragma unroll
    for (int ms = 0; ms < 4; ++ms) {
      v16bf afr = frag_a_16x32(&sA[buf][(wm * 64 + ms * 16) * TPAD], TPAD, lane);
#pragma unroll
      for (int ns = 0; ns < 2; ++ns)
        acc[ms][ns] = wmma_bf16(afr, bfr[ns], acc[ms][ns]);
    }
    wait_async_lds();
    __syncthreads();
  }
  const int b = bh >> 3, h = bh & 7;
  const int colLane = lane & 15;
  const int rquad = (lane >> 4) * 8;
#pragma unroll
  for (int ms = 0; ms < 4; ++ms)
#pragma unroll
    for (int ns = 0; ns < 2; ++ns) {
      const int col = n0 + wn * 32 + ns * 16 + colLane;   // d
#pragma unroll
      for (int i = 0; i < 8; ++i) {
        const int row = m0 + wm * 64 + ms * 16 + rquad + i;  // l
        ctx[((size_t)(b * LSEQ + row)) * D_MODEL + h * DH + col] =
            (__bf16)acc[ms][ns][i];
      }
    }
}

// ---------------------------------------------------------------------------

extern "C" void kernel_launch(void* const* d_in, const int* in_sizes, int n_in,
                              void* d_out, int out_size, void* d_ws, size_t ws_size,
                              hipStream_t stream) {
  (void)in_sizes; (void)n_in; (void)out_size; (void)ws_size;
  const float* x   = (const float*)d_in[0];
  const float* Wq  = (const float*)d_in[1];
  const float* bq  = (const float*)d_in[2];
  const float* Wk  = (const float*)d_in[3];
  const float* bk  = (const float*)d_in[4];
  const float* Wv  = (const float*)d_in[5];
  const float* bv  = (const float*)d_in[6];
  const float* Wo  = (const float*)d_in[7];
  const float* bo  = (const float*)d_in[8];
  const float* rel = (const float*)d_in[9];
  float* out = (float*)d_out;

  char* ws = (char*)d_ws;
  size_t off = 0;
  auto take = [&](size_t bytes) {
    char* p = ws + off;
    off += (bytes + 255) & ~(size_t)255;
    return p;
  };
  __bf16* xb   = (__bf16*)take((size_t)BL * D_MODEL * 2);
  __bf16* wqb  = (__bf16*)take((size_t)D_MODEL * D_MODEL * 2);
  __bf16* wkb  = (__bf16*)take((size_t)D_MODEL * D_MODEL * 2);
  __bf16* wvb  = (__bf16*)take((size_t)D_MODEL * D_MODEL * 2);
  __bf16* wob  = (__bf16*)take((size_t)D_MODEL * D_MODEL * 2);
  __bf16* relb = (__bf16*)take((size_t)(2 * 256 - 1) * D_MODEL * 2);
  __bf16* qb   = (__bf16*)take((size_t)BATCH * NH * LSEQ * DH * 2);
  __bf16* kb   = (__bf16*)take((size_t)BATCH * NH * LSEQ * DH * 2);
  __bf16* vb   = (__bf16*)take((size_t)BATCH * NH * LSEQ * DH * 2);
  float*  sc   = (float*) take((size_t)BATCH * NH * LSEQ * LSEQ * 4);
  __bf16* pb   = (__bf16*)take((size_t)BATCH * NH * LSEQ * LSEQ * 2);
  __bf16* cb   = (__bf16*)take((size_t)BL * D_MODEL * 2);

  // 1. convert everything to bf16
  cvt_f32_to_bf16<<<2048, 256, 0, stream>>>(x,   xb,  BL * D_MODEL);
  cvt_f32_to_bf16<<<2048, 256, 0, stream>>>(Wq,  wqb, D_MODEL * D_MODEL);
  cvt_f32_to_bf16<<<2048, 256, 0, stream>>>(Wk,  wkb, D_MODEL * D_MODEL);
  cvt_f32_to_bf16<<<2048, 256, 0, stream>>>(Wv,  wvb, D_MODEL * D_MODEL);
  cvt_f32_to_bf16<<<2048, 256, 0, stream>>>(Wo,  wob, D_MODEL * D_MODEL);
  cvt_f32_to_bf16<<<2048, 256, 0, stream>>>(rel, relb, (2 * 256 - 1) * D_MODEL);

  // 2. q/k/v projections (head-major bf16 outputs)
  dim3 gProj(16, 16, 1);
  gemm_bt<<<gProj, 256, 0, stream>>>(xb, 0u, wqb, 0u, D_MODEL, D_MODEL, bq, 0, qb, nullptr);
  gemm_bt<<<gProj, 256, 0, stream>>>(xb, 0u, wkb, 0u, D_MODEL, D_MODEL, bk, 0, kb, nullptr);
  gemm_bt<<<gProj, 256, 0, stream>>>(xb, 0u, wvb, 0u, D_MODEL, D_MODEL, bv, 0, vb, nullptr);

  // 3. content scores (mode 2), 4. rel scores (accumulate)
  dim3 gAtt(2, 2, BATCH * NH);
  gemm_bt<<<gAtt, 256, 0, stream>>>(qb, (unsigned)(LSEQ * DH), kb, (unsigned)(LSEQ * DH),
                                    DH, DH, nullptr, 2, nullptr, sc);
  attn_rel<<<NH * LSEQ, 128, 0, stream>>>(qb, relb, sc);

  // 5. softmax -> bf16 probabilities
  softmax_rows<<<(BATCH * NH * LSEQ) / 8, 256, 0, stream>>>(sc, pb);

  // 6. context = attn @ v (bf16 out, (B,L,D) layout)
  attn_ctx<<<gAtt, 256, 0, stream>>>(pb, vb, cb);

  // 7. output projection -> fp32 d_out
  gemm_bt<<<gProj, 256, 0, stream>>>(cb, 0u, wob, 0u, D_MODEL, D_MODEL, bo, 1, nullptr, out);
}
